// TextureWarpingModule_65171833749930
// MI455X (gfx1250) — matrix operand order repaired and checked by reading, not compiled, not run on hardware
//
#include <hip/hip_runtime.h>
#include <hip/hip_bf16.h>

// ---------------------------------------------------------------------------
// MI455X (gfx1250): all large convs as bf16 WMMA GEMMs with 4x M register
// blocking (4 accumulators/wave -> 4 wmma per B fetch), NHWC bf16 activations
// (B lane loads = 16 contiguous bf16), f32 accumulate. The 1x1 GEMMs stage the
// block-shared B tile into LDS with the Tensor Data Mover (TENSOR_LOAD_TO_LDS,
// double-buffered, s_wait_tensorcnt) when the builtin is available.
// ---------------------------------------------------------------------------

typedef __attribute__((ext_vector_type(16))) __bf16 bf16x16;
typedef __attribute__((ext_vector_type(8)))  __bf16 bf16x8;
typedef __attribute__((ext_vector_type(8)))  float  f32x8;
typedef __attribute__((ext_vector_type(4)))  unsigned int u32x4;
typedef __attribute__((ext_vector_type(8)))  int i32x8;
typedef __attribute__((ext_vector_type(4)))  int i32x4;

union AV { bf16x16 v; bf16x8 h[2]; __bf16 e[16]; };
union CV { f32x8 v; float e[8]; };

#define BATCH 8
#define CH    256
#define HW    64
#define SPAT  4096            // 64*64
#define NPOS  32768           // BATCH*SPAT
#define NTIL  2048            // NPOS/16

#if defined(__has_builtin)
#if __has_builtin(__builtin_amdgcn_tensor_load_to_lds) && \
    __has_builtin(__builtin_amdgcn_s_wait_tensorcnt)
#define HAVE_TDM 1
#endif
#endif
#ifndef HAVE_TDM
#define HAVE_TDM 0
#endif

#if HAVE_TDM
// TDM load of a 2D tile: 16 rows x 32 bf16 (64B/row), row pitch = K elements,
// into LDS at byte offset lds_off (contiguous 1KB). D# layout per CDNA5 ISA
// ch.8: group0 = {count/type/addr}, group1 = {data_size/dims/tile/stride}.
__device__ __forceinline__ void tdm_load_b_tile(const __bf16* gsrc, unsigned lds_off, int K)
{
    unsigned long long ga = (unsigned long long)(size_t)gsrc;
    u32x4 g0 = {
        1u,                                                  // count=1, user D#
        lds_off,                                             // lds_addr (bytes)
        (unsigned)(ga & 0xFFFFFFFFu),                        // global_addr[31:0]
        ((unsigned)(ga >> 32) & 0x01FFFFFFu) | (2u << 30)    // addr[56:32] | type=2
    };
    i32x8 g1 = {
        (1 << 16),                     // workgroup_mask=0, data_size=1 (2 bytes)
        0,                             // barrier addr 0, tensor_dim0 lo16 = 0
        0x4000,                        // tensor_dim0 = 1<<30 (hi16), dim1 lo16 = 0
        0x4000 | (32 << 16),           // tensor_dim1 = 1<<30 (hi16), tile_dim0 = 32
        16,                            // tile_dim1 = 16, tile_dim2 = 0
        K,                             // tensor_dim0_stride low32 (elements)
        0, 0                           // stride hi16 / dim1_stride (unused, 2D)
    };
    i32x4 z4 = { 0, 0, 0, 0 };
#if __clang_major__ >= 23
    i32x8 z8 = { 0, 0, 0, 0, 0, 0, 0, 0 };
    __builtin_amdgcn_tensor_load_to_lds(g0, g1, z4, z4, z8, 0);
#else
    __builtin_amdgcn_tensor_load_to_lds(g0, g1, z4, z4, 0);
#endif
}
#endif

__device__ __forceinline__ float dsigmoid(float x) { return 1.0f / (1.0f + __expf(-x)); }

// ---------------------------------------------------------------------------
// 1x1 conv as GEMM: Y[p,m] = bias[m] + sum_k A[m,k]*X[p,k].
// Block = 4 waves = entire M (4 M-groups x 4 tiles). blockIdx = N tile.
// B tile (16 cols x 32 k = 1KB) TDM-staged to LDS by wave 0, double-buffered.
// ---------------------------------------------------------------------------
__global__ void __launch_bounds__(128)
gemm1x1_wmma(const __bf16* __restrict__ A, const __bf16* __restrict__ X,
             const float* __restrict__ bias, float* __restrict__ Y,
             int K, int Mout)
{
    __shared__ __bf16 lbs[2][512];       // two 1KB B-tile buffers (LDS offs 0, 1024)

    int wave = threadIdx.x >> 5;
    int lane = threadIdx.x & 31;
    size_t n0 = (size_t)blockIdx.x << 4;
    int m0 = wave << 6;                  // this wave's 64 M rows
    int r  = lane & 15;
    int hi = lane >> 4;
    int khalf = hi << 4;

    const __bf16* bsrc = X + n0 * K;     // row-pitch-K 2D source for TDM

    const __bf16* ar[4];
#pragma unroll
    for (int t = 0; t < 4; ++t)
        ar[t] = A + (size_t)(m0 + (t << 4) + r) * K + khalf;

    CV acc[4];
#pragma unroll
    for (int t = 0; t < 4; ++t)
#pragma unroll
        for (int j = 0; j < 8; ++j) acc[t].e[j] = 0.0f;

#if HAVE_TDM
    if (wave == 0) {
        tdm_load_b_tile(bsrc, 0u, K);
        __builtin_amdgcn_s_wait_tensorcnt(0);
    }
    __syncthreads();
#endif

    for (int kb = 0; kb < K; kb += 32) {
        AV b;
#if HAVE_TDM
        int cur = (kb >> 5) & 1;
        if (wave == 0 && kb + 32 < K)    // async-prefetch next B tile via TDM
            tdm_load_b_tile(bsrc + kb + 32, (unsigned)((cur ^ 1) << 10), K);
        b.v = *(const bf16x16*)(&lbs[cur][0] + (r << 5) + (hi << 4));
#else
        b.v = *(const bf16x16*)(X + (n0 + r) * K + (hi << 4) + kb);
#endif
#pragma unroll
        for (int t = 0; t < 4; ++t) {
            __builtin_prefetch(ar[t] + kb + 256, 0, 0);
            AV a;
            a.h[0] = *(const bf16x8*)(ar[t] + kb);
            a.h[1] = *(const bf16x8*)(ar[t] + kb + 8);
            acc[t].v = __builtin_amdgcn_wmma_f32_16x16x32_bf16(
                false, a.v, false, b.v, (short)0, acc[t].v, false, false);
        }
#if HAVE_TDM
        if (wave == 0) __builtin_amdgcn_s_wait_tensorcnt(0);
        __syncthreads();
#endif
    }

    size_t n = n0 + r;
#pragma unroll
    for (int t = 0; t < 4; ++t) {
        int mbase = m0 + (t << 4) + (hi << 3);
#pragma unroll
        for (int j = 0; j < 8; ++j) {
            int m = mbase + j;
            if (m < Mout) Y[n * (size_t)Mout + m] = acc[t].e[j] + bias[m];
        }
    }
}

// ---------------------------------------------------------------------------
// 3x3 conv (pad 1) as gather-GEMM. K = 9*256 = 2304 (K = tap*256 + ci).
// 4x M blocking: one gathered B tile feeds 4 WMMAs. A rows padded to 256.
// ---------------------------------------------------------------------------
__global__ void __launch_bounds__(128)
gemmc3_wmma(const __bf16* __restrict__ A, const __bf16* __restrict__ X,
            const float* __restrict__ bias, float* __restrict__ Y, int Mout)
{
    const int K = 2304;
    int wave = threadIdx.x >> 5;
    int lane = threadIdx.x & 31;
    int job = blockIdx.x * 4 + wave;
    int mg = job & 3;                    // M group (64 rows)
    int tn = job >> 2;                   // N tile (shared by waves in block)
    int m0 = mg << 6;
    int r  = lane & 15;
    int hi = lane >> 4;
    int khalf = hi << 4;
    int kb2o  = hi << 4;

    int p = tn * 16 + r;
    int b = p >> 12, s = p & 4095, y = s >> 6, x = s & 63;

    const __bf16* ar[4];
#pragma unroll
    for (int t = 0; t < 4; ++t)
        ar[t] = A + (size_t)(m0 + (t << 4) + r) * K + khalf;

    CV acc[4];
#pragma unroll
    for (int t = 0; t < 4; ++t)
#pragma unroll
        for (int j = 0; j < 8; ++j) acc[t].e[j] = 0.0f;

    for (int kb = 0; kb < K; kb += 32) {
        int kb2 = kb + kb2o;
        int tap = kb2 >> 8;              // 0..8
        int ci0 = kb2 & 255;             // 16-aligned -> stays within one tap
        int yy = y + (tap / 3) - 1;
        int xx = x + (tap % 3) - 1;
        AV bm;
        if ((unsigned)yy < (unsigned)HW && (unsigned)xx < (unsigned)HW) {
            bm.v = *(const bf16x16*)(X + (((size_t)((b << 12) + (yy << 6) + xx)) << 8) + ci0);
        } else {
#pragma unroll
            for (int i = 0; i < 16; ++i) bm.e[i] = (__bf16)0.0f;
        }
#pragma unroll
        for (int t = 0; t < 4; ++t) {
            __builtin_prefetch(ar[t] + kb + 256, 0, 0);
            AV a;
            a.h[0] = *(const bf16x8*)(ar[t] + kb);
            a.h[1] = *(const bf16x8*)(ar[t] + kb + 8);
            acc[t].v = __builtin_amdgcn_wmma_f32_16x16x32_bf16(
                false, a.v, false, bm.v, (short)0, acc[t].v, false, false);
        }
    }

    size_t n = (size_t)tn * 16 + r;
#pragma unroll
    for (int t = 0; t < 4; ++t) {
        int mbase = m0 + (t << 4) + (hi << 3);
#pragma unroll
        for (int j = 0; j < 8; ++j) {
            int m = mbase + j;
            if (m < Mout) Y[n * (size_t)Mout + m] = acc[t].e[j] + bias[m];
        }
    }
}

// ---------------------------------------------------------------------------
// Deformable conv (dg=8, 3x3, pad 1) as sample-GEMM, K = (g*9+k)*32 + cpg.
// One bilinear-sampled, mask-weighted B tile feeds 4 WMMAs (4x M blocking).
// Output NCHW f32 (warp).
// ---------------------------------------------------------------------------
__global__ void __launch_bounds__(128)
gemmdcn_wmma(const __bf16* __restrict__ A, const __bf16* __restrict__ Xm,
             const float* __restrict__ co, const float* __restrict__ bias,
             float* __restrict__ Out)
{
    const int K = 2304;
    int wave = threadIdx.x >> 5;
    int lane = threadIdx.x & 31;
    int job = blockIdx.x * 4 + wave;
    int mg = job & 3;
    int tn = job >> 2;
    int m0 = mg << 6;
    int r  = lane & 15;
    int hi = lane >> 4;
    int khalf = hi << 4;
    int kb2o  = hi << 4;

    int p = tn * 16 + r;
    int b = p >> 12, s = p & 4095, y = s >> 6, x = s & 63;
    float yf = (float)y, xf = (float)x;
    const float* cr = co + (size_t)p * 216;

    const __bf16* ar[4];
#pragma unroll
    for (int t = 0; t < 4; ++t)
        ar[t] = A + (size_t)(m0 + (t << 4) + r) * K + khalf;

    CV acc[4];
#pragma unroll
    for (int t = 0; t < 4; ++t)
#pragma unroll
        for (int j = 0; j < 8; ++j) acc[t].e[j] = 0.0f;

    for (int kb = 0; kb < K; kb += 32) {
        int kb2 = kb + kb2o;
        int gk   = kb2 >> 5;             // g*9 + k  (0..71)
        int cpg0 = kb2 & 31;             // 0 or 16
        int g  = gk / 9;
        int k9 = gk - g * 9;
        float py = yf + (float)(k9 / 3) - 1.0f + cr[g * 18 + 2 * k9];
        float px = xf + (float)(k9 % 3) - 1.0f + cr[g * 18 + 2 * k9 + 1];
        float mval = dsigmoid(cr[144 + gk]);
        float fy0 = floorf(py), fx0 = floorf(px);
        float ly = py - fy0, lx = px - fx0;
        int iy0 = (int)fy0, ix0 = (int)fx0;
        int cb = (g << 5) + cpg0;

        float sf[16];
#pragma unroll
        for (int i = 0; i < 16; ++i) sf[i] = 0.0f;
#pragma unroll
        for (int cy = 0; cy < 2; ++cy) {
            int iy = iy0 + cy;
            float wy = cy ? ly : (1.0f - ly);
            if ((unsigned)iy < (unsigned)HW) {
#pragma unroll
                for (int cx = 0; cx < 2; ++cx) {
                    int ix = ix0 + cx;
                    float wgt = wy * (cx ? lx : (1.0f - lx));
                    if ((unsigned)ix < (unsigned)HW) {
                        AV vv;
                        vv.v = *(const bf16x16*)(Xm + (((size_t)((b << 12) + (iy << 6) + ix)) << 8) + cb);
#pragma unroll
                        for (int i = 0; i < 16; ++i) sf[i] += wgt * (float)vv.e[i];
                    }
                }
            }
        }

        AV bm;
#pragma unroll
        for (int i = 0; i < 16; ++i) bm.e[i] = (__bf16)(mval * sf[i]);
#pragma unroll
        for (int t = 0; t < 4; ++t) {
            AV a;
            a.h[0] = *(const bf16x8*)(ar[t] + kb);
            a.h[1] = *(const bf16x8*)(ar[t] + kb + 8);
            acc[t].v = __builtin_amdgcn_wmma_f32_16x16x32_bf16(
                false, a.v, false, bm.v, (short)0, acc[t].v, false, false);
        }
    }

#pragma unroll
    for (int t = 0; t < 4; ++t) {
        int mbase = m0 + (t << 4) + (hi << 3);
#pragma unroll
        for (int j = 0; j < 8; ++j) {
            int m = mbase + j;
            Out[(((size_t)b << 8) + m) * SPAT + s] = acc[t].e[j] + bias[m];
        }
    }
}

// ---------------------------------------------------------------------------
// Depthwise 7x7 (pad 3), NHWC bf16 in, NHWC f32 out. Memory-bound VALU job.
// ---------------------------------------------------------------------------
__global__ void dwconv7(const __bf16* __restrict__ X, const float* __restrict__ w2,
                        const float* __restrict__ b2, float* __restrict__ Y)
{
    size_t idx = (size_t)blockIdx.x * blockDim.x + threadIdx.x;   // < NPOS*256
    int c = (int)(idx & 255);
    int p = (int)(idx >> 8);
    int b = p >> 12, s = p & 4095, y = s >> 6, x = s & 63;
    float acc = b2[c];
    const float* wc = w2 + c * 49;
    for (int ky = 0; ky < 7; ++ky) {
        int yy = y + ky - 3;
        if ((unsigned)yy >= (unsigned)HW) continue;
        for (int kx = 0; kx < 7; ++kx) {
            int xx = x + kx - 3;
            if ((unsigned)xx >= (unsigned)HW) continue;
            acc += wc[ky * 7 + kx] *
                   (float)X[(((size_t)((b << 12) + (yy << 6) + xx)) << 8) + c];
        }
    }
    Y[idx] = acc;
}

// ---------------------------------------------------------------------------
// GroupNorm (32 groups, 8 ch/group): per (batch, group) mean/rstd reduction.
// ---------------------------------------------------------------------------
__global__ void gn_stats(const float* __restrict__ Y, float* __restrict__ stats)
{
    __shared__ float ssum[256];
    __shared__ float ssq[256];
    int bg = blockIdx.x;            // b*32 + group
    int b = bg >> 5, gr = bg & 31;
    int tid = threadIdx.x;
    float sum = 0.0f, sq = 0.0f;
    for (int i = tid; i < 8 * SPAT; i += 256) {
        int sp = i >> 3, cj = i & 7;
        float v = Y[(((size_t)b * SPAT + sp) << 8) + (gr << 3) + cj];
        sum += v; sq += v * v;
    }
    ssum[tid] = sum; ssq[tid] = sq;
    __syncthreads();
    for (int st = 128; st > 0; st >>= 1) {
        if (tid < st) { ssum[tid] += ssum[tid + st]; ssq[tid] += ssq[tid + st]; }
        __syncthreads();
    }
    if (tid == 0) {
        float m = ssum[0] * (1.0f / (8.0f * SPAT));
        float v = ssq[0] * (1.0f / (8.0f * SPAT)) - m * m;
        stats[bg * 2]     = m;
        stats[bg * 2 + 1] = rsqrtf(v + 1e-6f);
    }
}

__global__ void gn_apply_silu(const float* __restrict__ Y, const float* __restrict__ stats,
                              const float* __restrict__ gamma, const float* __restrict__ beta,
                              __bf16* __restrict__ Xout, float* __restrict__ OutNCHW)
{
    size_t idx = (size_t)blockIdx.x * blockDim.x + threadIdx.x;   // < NPOS*256
    int c = (int)(idx & 255);
    int p = (int)(idx >> 8);
    int b = p >> 12, gr = c >> 3;
    float m = stats[(b * 32 + gr) * 2];
    float r = stats[(b * 32 + gr) * 2 + 1];
    float v = (Y[idx] - m) * r * gamma[c] + beta[c];
    float a = v * dsigmoid(v);                     // SiLU
    Xout[idx] = (__bf16)a;
    if (OutNCHW) OutNCHW[(((size_t)b << 8) + c) * SPAT + (p & 4095)] = a;
}

__global__ void cvt_bf16(const float* __restrict__ Y, __bf16* __restrict__ Xout)
{
    size_t idx = (size_t)blockIdx.x * blockDim.x + threadIdx.x;
    Xout[idx] = (__bf16)Y[idx];
}

// ---------------------------------------------------------------------------
// Input packing: bilinear-upsample cond 16->64 (half-pixel centers, clamped)
// and build concat activation X0 [NPOS][512] bf16 + main NHWC Xm [NPOS][256].
// ---------------------------------------------------------------------------
__global__ void pack_inputs(const float* __restrict__ mainf, const float* __restrict__ cond,
                            __bf16* __restrict__ X0, __bf16* __restrict__ Xm)
{
    size_t idx = (size_t)blockIdx.x * blockDim.x + threadIdx.x;   // < NPOS*256
    int c = (int)(idx & 255);
    int p = (int)(idx >> 8);
    int b = p >> 12, s = p & 4095, y = s >> 6, x = s & 63;

    float mv = mainf[(((size_t)b << 8) + c) * SPAT + s];
    Xm[idx] = (__bf16)mv;
    X0[(size_t)p * 512 + 256 + c] = (__bf16)mv;

    float sy = ((float)y + 0.5f) * 0.25f - 0.5f;
    float sx = ((float)x + 0.5f) * 0.25f - 0.5f;
    float fy = floorf(sy), fx = floorf(sx);
    float ty = sy - fy, tx = sx - fx;
    int y0 = (int)fy, x0 = (int)fx;
    int y0c = min(15, max(0, y0)),     x0c = min(15, max(0, x0));
    int y1c = min(15, max(0, y0 + 1)), x1c = min(15, max(0, x0 + 1));
    const float* cp = cond + (((size_t)b << 8) + c) * 256;   // 16*16
    float v00 = cp[y0c * 16 + x0c], v01 = cp[y0c * 16 + x1c];
    float v10 = cp[y1c * 16 + x0c], v11 = cp[y1c * 16 + x1c];
    float val = (1.0f - ty) * ((1.0f - tx) * v00 + tx * v01)
              +         ty  * ((1.0f - tx) * v10 + tx * v11);
    X0[(size_t)p * 512 + c] = (__bf16)val;
}

// ---------------------------------------------------------------------------
// Weight packing kernels (f32 -> bf16 in WMMA-friendly K layouts).
// ---------------------------------------------------------------------------
__global__ void pack_w_dense(const float* __restrict__ w, __bf16* __restrict__ out, int n)
{
    int i = blockIdx.x * blockDim.x + threadIdx.x;
    if (i < n) out[i] = (__bf16)w[i];
}

// w: [M][256][3][3]  ->  out: [Mpad][2304] with K = tap*256 + ci (zero-pad rows >= M)
__global__ void pack_w_conv3(const float* __restrict__ w, __bf16* __restrict__ out,
                             int M, int n)
{
    int i = blockIdx.x * blockDim.x + threadIdx.x;
    if (i >= n) return;
    int m = i / 2304, kk = i - m * 2304;
    int tap = kk >> 8, ci = kk & 255;
    out[i] = (m < M) ? (__bf16)w[((size_t)m * 256 + ci) * 9 + tap] : (__bf16)0.0f;
}

// w_dcn: [256][256][3][3] -> out: [256][2304] with K = (g*9+k)*32 + cpg, ci = g*32+cpg
__global__ void pack_w_dcn(const float* __restrict__ w, __bf16* __restrict__ out)
{
    int i = blockIdx.x * blockDim.x + threadIdx.x;
    if (i >= 256 * 2304) return;
    int m = i / 2304, kk = i - m * 2304;
    int gk = kk >> 5, cpg = kk & 31;
    int g = gk / 9, k = gk - g * 9;
    int ci = (g << 5) + cpg;
    out[i] = (__bf16)w[((size_t)m * 256 + ci) * 9 + k];
}

// ---------------------------------------------------------------------------
// Host-side orchestration.
// ---------------------------------------------------------------------------
extern "C" void kernel_launch(void* const* d_in, const int* in_sizes, int n_in,
                              void* d_out, int out_size, void* d_ws, size_t ws_size,
                              hipStream_t stream)
{
    (void)in_sizes; (void)n_in; (void)out_size; (void)ws_size;

    const float* mainf = (const float*)d_in[0];
    const float* cond  = (const float*)d_in[1];
    const float* w1  = (const float*)d_in[2];
    const float* b1  = (const float*)d_in[3];
    const float* g1  = (const float*)d_in[4];
    const float* be1 = (const float*)d_in[5];
    const float* w2  = (const float*)d_in[6];
    const float* b2  = (const float*)d_in[7];
    const float* g2  = (const float*)d_in[8];
    const float* be2 = (const float*)d_in[9];
    const float* w3  = (const float*)d_in[10];
    const float* b3  = (const float*)d_in[11];
    const float* w4  = (const float*)d_in[12];
    const float* b4  = (const float*)d_in[13];
    const float* g3  = (const float*)d_in[14];
    const float* be3 = (const float*)d_in[15];
    const float* wco = (const float*)d_in[16];
    const float* bco = (const float*)d_in[17];
    const float* wdc = (const float*)d_in[18];
    const float* bdc = (const float*)d_in[19];

    char* ws = (char*)d_ws;
    size_t off = 0;
    auto carve = [&](size_t bytes) -> char* {
        char* q = ws + off;
        off += (bytes + 255) & ~(size_t)255;
        return q;
    };
    __bf16* X0   = (__bf16*)carve((size_t)NPOS * 512 * 2);   // concat activation
    __bf16* Xm   = (__bf16*)carve((size_t)NPOS * 256 * 2);   // main NHWC bf16
    __bf16* Xa   = (__bf16*)carve((size_t)NPOS * 256 * 2);   // stage activation (reused)
    float*  Yf   = (float*) carve((size_t)NPOS * 256 * 4);   // f32 GEMM out (reused)
    float*  Co   = (float*) carve((size_t)NPOS * 216 * 4);   // offsets+mask raw
    __bf16* w1b  = (__bf16*)carve((size_t)256 * 512 * 2);
    __bf16* w3b  = (__bf16*)carve((size_t)256 * 256 * 2);
    __bf16* w4b  = (__bf16*)carve((size_t)256 * 2304 * 2);
    __bf16* wcob = (__bf16*)carve((size_t)256 * 2304 * 2);   // padded 216 -> 256 rows
    __bf16* wdbn = (__bf16*)carve((size_t)256 * 2304 * 2);
    float*  st   = (float*) carve((size_t)BATCH * 32 * 2 * 4);

    float* warp_out = (float*)d_out;                               // output 0 (NCHW)
    float* feat_out = (float*)d_out + (size_t)NPOS * 256;          // output 1 (NCHW)

    const int EW_BLOCKS = (NPOS * 256) / 256;    // 32768 blocks of 256 threads

    // --- weight packing ---
    pack_w_dense<<<(256 * 512 + 255) / 256, 256, 0, stream>>>(w1, w1b, 256 * 512);
    pack_w_dense<<<(256 * 256 + 255) / 256, 256, 0, stream>>>(w3, w3b, 256 * 256);
    pack_w_conv3<<<(256 * 2304 + 255) / 256, 256, 0, stream>>>(w4, w4b, 256, 256 * 2304);
    pack_w_conv3<<<(256 * 2304 + 255) / 256, 256, 0, stream>>>(wco, wcob, 216, 256 * 2304);
    pack_w_dcn  <<<(256 * 2304 + 255) / 256, 256, 0, stream>>>(wdc, wdbn);

    // --- input packing (bilinear upsample + concat + main NHWC) ---
    pack_inputs<<<EW_BLOCKS, 256, 0, stream>>>(mainf, cond, X0, Xm);

    // --- stage 1: 1x1 conv (512 -> 256), GN, SiLU ---
    gemm1x1_wmma<<<NTIL, 128, 0, stream>>>(w1b, X0, b1, Yf, 512, 256);
    gn_stats<<<BATCH * 32, 256, 0, stream>>>(Yf, st);
    gn_apply_silu<<<EW_BLOCKS, 256, 0, stream>>>(Yf, st, g1, be1, Xa, nullptr);

    // --- stage 2: depthwise 7x7, GN, SiLU ---
    dwconv7<<<EW_BLOCKS, 256, 0, stream>>>(Xa, w2, b2, Yf);
    gn_stats<<<BATCH * 32, 256, 0, stream>>>(Yf, st);
    gn_apply_silu<<<EW_BLOCKS, 256, 0, stream>>>(Yf, st, g2, be2, Xa, nullptr);

    // --- stage 3: 1x1 conv (256 -> 256), no norm ---
    gemm1x1_wmma<<<NTIL, 128, 0, stream>>>(w3b, Xa, b3, Yf, 256, 256);
    cvt_bf16<<<EW_BLOCKS, 256, 0, stream>>>(Yf, Xa);

    // --- stage 4: 3x3 conv (256 -> 256), GN, SiLU -> offset_feat (output 1) ---
    gemmc3_wmma<<<NTIL, 128, 0, stream>>>(w4b, Xa, b4, Yf, 256);
    gn_stats<<<BATCH * 32, 256, 0, stream>>>(Yf, st);
    gn_apply_silu<<<EW_BLOCKS, 256, 0, stream>>>(Yf, st, g3, be3, Xa, feat_out);

    // --- stage 5: conv_offset 3x3 (256 -> 216, padded to 256 rows) ---
    gemmc3_wmma<<<NTIL, 128, 0, stream>>>(wcob, Xa, bco, Co, 216);

    // --- stage 6: deformable conv -> warp (output 0) ---
    gemmdcn_wmma<<<NTIL, 128, 0, stream>>>(wdbn, Xm, Co, bdc, warp_out);
}
